// CART_69355131895963
// MI455X (gfx1250) — matrix operand order repaired and checked by reading, not compiled, not run on hardware
//
#include <hip/hip_runtime.h>

typedef _Float16 h16;
typedef __attribute__((ext_vector_type(16))) _Float16 v16h;
typedef __attribute__((ext_vector_type(8)))  _Float16 v8h;
typedef __attribute__((ext_vector_type(4)))  _Float16 v4h;
typedef __attribute__((ext_vector_type(8)))  float    v8f;

#define NB     65536   // batch
#define NF     256     // features
#define NS     32      // slots per tree
#define NO     16      // outputs
#define NT     32      // trees
#define TILE_M 256     // rows per block
#define XPITCH 264     // f16 pitch for xn tile (16B aligned rows, bank-staggered)
#define SPITCH 40      // f16 pitch for score/o1 staging (80B rows; also float pitch 20)

union AF { v16h v; v8h h[2]; };

// ---------------------------------------------------------------------------
// 1) BatchNorm statistics: per-block partial column sums (coalesced rows)
// ---------------------------------------------------------------------------
__global__ void k_bn_partial(const float* __restrict__ x,
                             float* __restrict__ psum, float* __restrict__ psq) {
  int f = threadIdx.x;                       // feature column
  size_t rb = (size_t)blockIdx.x * 256;      // row chunk
  const float* p = x + rb * NF + f;
  float s = 0.f, q = 0.f;
  #pragma unroll 4
  for (int r = 0; r < 256; ++r) { float v = p[(size_t)r * NF]; s += v; q += v * v; }
  psum[blockIdx.x * NF + f] = s;
  psq [blockIdx.x * NF + f] = q;
}

__global__ void k_bn_final(const float* __restrict__ psum, const float* __restrict__ psq,
                           const float* __restrict__ gamma, const float* __restrict__ beta,
                           float* __restrict__ scale, float* __restrict__ shift) {
  int f = threadIdx.x;
  float s = 0.f, q = 0.f;
  for (int j = 0; j < 256; ++j) { s += psum[j * NF + f]; q += psq[j * NF + f]; }
  float mean = s * (1.0f / NB);
  float var  = q * (1.0f / NB) - mean * mean;   // biased variance
  float sc   = gamma[f] * rsqrtf(var + 1e-5f);
  scale[f] = sc;
  shift[f] = beta[f] - mean * sc;
}

// ---------------------------------------------------------------------------
// 2) sparsemax along F for each (tree, slot): Michelot fixed-point projection.
//    One wave32 per column of 256; writes P as f16 [T][F][S] (B-operand ready).
// ---------------------------------------------------------------------------
__global__ void k_sparsemax(const float* __restrict__ fsm, h16* __restrict__ Ph) {
  int lane = threadIdx.x & 31;
  int wid  = threadIdx.x >> 5;
  int col  = blockIdx.x * 8 + wid;           // 0..1023
  int t = col >> 5, s = col & 31;
  const float* base = fsm + (size_t)t * NF * NS + s;
  float z[8];
  #pragma unroll
  for (int j = 0; j < 8; ++j) z[j] = base[(size_t)(lane + 32 * j) * NS];

  float sum = 0.f;
  #pragma unroll
  for (int j = 0; j < 8; ++j) sum += z[j];
  #pragma unroll
  for (int o = 16; o > 0; o >>= 1) sum += __shfl_xor(sum, o, 32);

  int   cnt = NF;
  float tau = (sum - 1.0f) / (float)cnt;
  for (int it = 0; it < NF; ++it) {
    float ns = 0.f; int nc = 0;
    #pragma unroll
    for (int j = 0; j < 8; ++j) if (z[j] > tau) { ns += z[j]; nc++; }
    #pragma unroll
    for (int o = 16; o > 0; o >>= 1) { ns += __shfl_xor(ns, o, 32); nc += __shfl_xor(nc, o, 32); }
    if (nc == cnt) break;                    // active set stable -> tau is fixed point
    cnt = nc;
    tau = (ns - 1.0f) / (float)nc;
  }
  h16* dst = Ph + (size_t)t * NF * NS + s;
  #pragma unroll
  for (int j = 0; j < 8; ++j) {
    float p = z[j] - tau;
    dst[(size_t)(lane + 32 * j) * NS] = (h16)(p > 0.f ? p : 0.f);
  }
}

// ---------------------------------------------------------------------------
// 2b) Pre-convert W1/W2 to f16 (B-fragment-ready row-major layouts)
// ---------------------------------------------------------------------------
__global__ void k_wconv(const float* __restrict__ W1, const float* __restrict__ W2,
                        h16* __restrict__ W1h, h16* __restrict__ W2h) {
  int i = blockIdx.x * 256 + threadIdx.x;
  if (i < NT * NS * NS) W1h[i] = (h16)W1[i];
  if (i < NT * NS * NO) W2h[i] = (h16)W2[i];
}

// ---------------------------------------------------------------------------
// 3) Fused forest kernel: xn tile in LDS (f16) -> A-frags hoisted to VGPRs,
//    loop 32 trees with WMMA chain fed from registers + L2
// ---------------------------------------------------------------------------
__global__ void __launch_bounds__(256) k_forest(
    const float* __restrict__ x,   const h16* __restrict__ Ph,
    const h16* __restrict__ W1h,   const h16* __restrict__ W2h,
    const float* __restrict__ scale, const float* __restrict__ shift,
    const float* __restrict__ cut, const float* __restrict__ b1,
    const float* __restrict__ b2,  const float* __restrict__ tw,
    float* __restrict__ out) {
  extern __shared__ char smem[];
  h16* sX = (h16*)smem;                              // [TILE_M][XPITCH] f16
  h16* sS = (h16*)(smem + TILE_M * XPITCH * 2);      // [TILE_M][SPITCH] f16 / [TILE_M][20] f32
  __shared__ float sScale[NF], sShift[NF];
  __shared__ float sCut[NT * NS], sB1[NT * NS], sB2[NT * NO], sTw[NT * NO];

  int tid = threadIdx.x;
  sScale[tid] = scale[tid];
  sShift[tid] = shift[tid];
  for (int i = tid; i < NT * NS; i += 256) { sCut[i] = cut[i]; sB1[i] = b1[i]; }
  for (int i = tid; i < NT * NO; i += 256) { sB2[i] = b2[i]; sTw[i] = tw[i]; }
  __syncthreads();

  // Stage normalized f16 tile: xn = x*scale + shift, float4 coalesced loads
  size_t rowbase = (size_t)blockIdx.x * TILE_M;
  for (int it = 0; it < 64; ++it) {
    int idx = it * 256 + tid;
    int row = idx >> 6;
    int c4  = (idx & 63) * 4;
    float4 v = *(const float4*)(x + (rowbase + row) * NF + c4);
    v4h h;
    h.x = (h16)(v.x * sScale[c4 + 0] + sShift[c4 + 0]);
    h.y = (h16)(v.y * sScale[c4 + 1] + sShift[c4 + 1]);
    h.z = (h16)(v.z * sScale[c4 + 2] + sShift[c4 + 2]);
    h.w = (h16)(v.w * sScale[c4 + 3] + sShift[c4 + 3]);
    *(v4h*)(sX + row * XPITCH + c4) = h;
  }
  __syncthreads();

  int lane = tid & 31;
  int wave = tid >> 5;
  int lo = lane & 15, hi = lane >> 4;
  int m0w = wave * 32;                      // each wave owns a 32-row stripe

  // Hoist the whole K=256 A-stripe into registers (2 Mtiles x 8 ksteps = 128 VGPRs);
  // the tree loop then never touches LDS for GEMM1.
  AF aReg[8][2];
  #pragma unroll
  for (int ks = 0; ks < 8; ++ks)
    #pragma unroll
    for (int mt = 0; mt < 2; ++mt) {
      const h16* src = sX + (m0w + mt * 16 + lo) * XPITCH + ks * 32 + 8 * hi;
      aReg[ks][mt].h[0] = *(const v8h*)(src);
      aReg[ks][mt].h[1] = *(const v8h*)(src + 16);
    }

  // Per-lane constant: sum_t b2[t][n]*tw[t][n]  (folds bias into final add)
  float cbase = 0.f;
  #pragma unroll
  for (int t = 0; t < NT; ++t) cbase += sB2[t * NO + lo] * sTw[t * NO + lo];

  float acc[2][8];
  #pragma unroll
  for (int mt = 0; mt < 2; ++mt)
    #pragma unroll
    for (int r = 0; r < 8; ++r) acc[mt][r] = 0.f;

  for (int t = 0; t < NT; ++t) {
    const h16* PhT = Ph + (size_t)t * NF * NS;
    if (t + 1 < NT) __builtin_prefetch(PhT + NF * NS + lane * 256, 0, 0);

    // --- GEMM1: xp = xn[256x256] @ P[t][256x32], K in 8 steps of 32 ---
    v8f c1[2][2] = {};
    #pragma unroll
    for (int ks = 0; ks < 8; ++ks) {
      int k0 = ks * 32;
      AF b[2];
      #pragma unroll
      for (int nt = 0; nt < 2; ++nt)
        for (int s = 0; s < 16; ++s) b[nt].v[s] = PhT[(size_t)(k0 + 8 * hi + (s & 7) + 16 * (s >> 3)) * NS + nt * 16 + lo];
      #pragma unroll
      for (int mt = 0; mt < 2; ++mt)
        #pragma unroll
        for (int nt = 0; nt < 2; ++nt)
          c1[mt][nt] = __builtin_amdgcn_wmma_f32_16x16x32_f16(
              false, aReg[ks][mt].v, false, b[nt].v, (short)0, c1[mt][nt], false, false);
    }

    // score = sigmoid(xp - cut); re-layout via LDS (C-layout -> A-layout)
    #pragma unroll
    for (int mt = 0; mt < 2; ++mt)
      #pragma unroll
      for (int nt = 0; nt < 2; ++nt)
        #pragma unroll
        for (int r = 0; r < 8; ++r) {
          int n = nt * 16 + lo;
          int m = m0w + mt * 16 + r + 8 * hi;
          float v = c1[mt][nt][r] - sCut[t * NS + n];
          v = 1.0f / (1.0f + __expf(-v));
          sS[m * SPITCH + n] = (h16)v;
        }

    // --- GEMM2: o1 = relu(score[256x32] @ W1[t][32x32] + b1), K=32 in one WMMA ---
    AF a2[2], bw[2];
    #pragma unroll
    for (int mt = 0; mt < 2; ++mt) {
      const h16* src = sS + (m0w + mt * 16 + lo) * SPITCH + 8 * hi;
      a2[mt].h[0] = *(const v8h*)(src);
      a2[mt].h[1] = *(const v8h*)(src + 16);
    }
    #pragma unroll
    for (int nt = 0; nt < 2; ++nt)
      for (int s = 0; s < 16; ++s) bw[nt].v[s] = W1h[((size_t)t * NS + 8 * hi + (s & 7) + 16 * (s >> 3)) * NS + nt * 16 + lo];
    v8f c2[2][2] = {};
    #pragma unroll
    for (int mt = 0; mt < 2; ++mt)
      #pragma unroll
      for (int nt = 0; nt < 2; ++nt)
        c2[mt][nt] = __builtin_amdgcn_wmma_f32_16x16x32_f16(
            false, a2[mt].v, false, bw[nt].v, (short)0, c2[mt][nt], false, false);

    #pragma unroll
    for (int mt = 0; mt < 2; ++mt)
      #pragma unroll
      for (int nt = 0; nt < 2; ++nt)
        #pragma unroll
        for (int r = 0; r < 8; ++r) {
          int n = nt * 16 + lo;
          int m = m0w + mt * 16 + r + 8 * hi;
          float v = c2[mt][nt][r] + sB1[t * NS + n];
          sS[m * SPITCH + n] = (h16)(v > 0.f ? v : 0.f);
        }

    // --- GEMM3: o2 = o1[256x32] @ W2[t][32x16]; acc += o2*tw (b2 folded) ---
    AF a3[2], bw2;
    #pragma unroll
    for (int mt = 0; mt < 2; ++mt) {
      const h16* src = sS + (m0w + mt * 16 + lo) * SPITCH + 8 * hi;
      a3[mt].h[0] = *(const v8h*)(src);
      a3[mt].h[1] = *(const v8h*)(src + 16);
    }
    for (int s = 0; s < 16; ++s) bw2.v[s] = W2h[((size_t)t * NS + 8 * hi + (s & 7) + 16 * (s >> 3)) * NO + lo];
    v8f c3[2] = {};
    #pragma unroll
    for (int mt = 0; mt < 2; ++mt)
      c3[mt] = __builtin_amdgcn_wmma_f32_16x16x32_f16(
          false, a3[mt].v, false, bw2.v, (short)0, c3[mt], false, false);

    float twn = sTw[t * NO + lo];
    #pragma unroll
    for (int mt = 0; mt < 2; ++mt)
      #pragma unroll
      for (int r = 0; r < 8; ++r)
        acc[mt][r] += c3[mt][r] * twn;
  }

  // Epilogue: mean over trees (+ folded bias term), coalesced float4 stores
  float* sO = (float*)sS;                    // [TILE_M][20] floats, same byte rows
  #pragma unroll
  for (int mt = 0; mt < 2; ++mt)
    #pragma unroll
    for (int r = 0; r < 8; ++r)
      sO[(m0w + mt * 16 + r + 8 * hi) * 20 + lo] = (acc[mt][r] + cbase) * (1.0f / NT);
  __syncthreads();
  #pragma unroll
  for (int i = 0; i < 4; ++i) {
    int idx = i * 256 + tid;
    int row = idx >> 2, seg = idx & 3;
    float4 v = *(const float4*)(sO + row * 20 + seg * 4);
    *(float4*)(out + (rowbase + row) * NO + seg * 4) = v;
  }
}

// ---------------------------------------------------------------------------
extern "C" void kernel_launch(void* const* d_in, const int* in_sizes, int n_in,
                              void* d_out, int out_size, void* d_ws, size_t ws_size,
                              hipStream_t stream) {
  (void)in_sizes; (void)n_in; (void)out_size; (void)ws_size;
  const float* x     = (const float*)d_in[0];
  const float* gamma = (const float*)d_in[1];
  const float* beta  = (const float*)d_in[2];
  const float* fsm   = (const float*)d_in[3];
  const float* cut   = (const float*)d_in[4];
  const float* W1    = (const float*)d_in[5];
  const float* b1    = (const float*)d_in[6];
  const float* W2    = (const float*)d_in[7];
  const float* b2    = (const float*)d_in[8];
  const float* tw    = (const float*)d_in[9];
  float* out = (float*)d_out;

  char* ws = (char*)d_ws;
  float* psum  = (float*)(ws);                  // 256*256 f32 = 256 KB
  float* psq   = (float*)(ws + 256 * 1024);     // 256 KB
  float* scale = (float*)(ws + 512 * 1024);     // 1 KB
  float* shift = (float*)(ws + 513 * 1024);     // 1 KB
  h16*   Ph    = (h16*)  (ws + 516 * 1024);     // 32*256*32 f16 = 512 KB
  h16*   W1h   = (h16*)  (ws + 1028 * 1024);    // 64 KB
  h16*   W2h   = (h16*)  (ws + 1092 * 1024);    // 32 KB

  k_bn_partial<<<256, 256, 0, stream>>>(x, psum, psq);
  k_bn_final  <<<1,   256, 0, stream>>>(psum, psq, gamma, beta, scale, shift);
  k_sparsemax <<<128, 256, 0, stream>>>(fsm, Ph);
  k_wconv     <<<128, 256, 0, stream>>>(W1, W2, W1h, W2h);

  size_t smem = (size_t)TILE_M * XPITCH * 2 + (size_t)TILE_M * SPITCH * 2; // ~152 KB
  k_forest<<<256, 256, smem, stream>>>(x, Ph, W1h, W2h, scale, shift,
                                       cut, b1, b2, tw, out);
}
